// DeformableSelfAttention_62783831933740
// MI455X (gfx1250) — compile-verified, matrix operand order checked by reading, not run on hardware
//
#include <hip/hip_runtime.h>

// ---------------------------------------------------------------------------
// DeformableSelfAttention for MI455X (gfx1250, wave32, WMMA)
//   B=2, C=256, H=W=64, HW=4096, head dim 32.
// All GEMM-shaped math on v_wmma_f32_16x16x32_bf16; operands fed by 16-byte
// contiguous loads; softmax row reductions done via an LDS transpose (2
// shuffles per 64-key chunk) instead of per-row ds_bpermute butterflies.
// ---------------------------------------------------------------------------

#define CIN   256
#define HDIM  64
#define WDIM  64
#define HW    4096
#define BATCH 2
#define KTAPS 9
#define RDIM  (CIN * KTAPS)   // 2304, K-dim of the deform-conv GEMMs
#define C8    32              // q/k channels

typedef __attribute__((ext_vector_type(16))) __bf16 v16bf;
typedef __attribute__((ext_vector_type(8)))  float  v8f;

union BV { v16bf v; unsigned short h[16]; uint4 q[2]; };
union V8 { v8f v; float4 f4[2]; float f[8]; };

__device__ __forceinline__ unsigned short f2bf(float f) {
  union { float f; unsigned u; } c; c.f = f;
  unsigned u = c.u + 0x7FFFu + ((c.u >> 16) & 1u);   // round-to-nearest-even
  return (unsigned short)(u >> 16);
}

// bf16 operand loader for wmma_*_16x16x32 (wave32), K contiguous in memory.
// "outer" (A: row m / B: col n) is lane-striped with stride `so`; per ISA
// 7.12.2 lanes 0-15 hold K {0..7, 16..23}, lanes 16-31 hold K {8..15, 24..31}.
// Requires 16-byte alignment of base + outer*so + i0 (+8).
__device__ __forceinline__ v16bf load_row16(const unsigned short* base, long so,
                                            int o0, int i0, int lane) {
  const unsigned short* p =
      base + (long)(o0 + (lane & 15)) * so + i0 + 8 * (lane >> 4);
  BV r;
  r.q[0] = *(const uint4*)(p);        // K {i0..i0+7}   (+8 for hi half)
  r.q[1] = *(const uint4*)(p + 16);   // K {i0+16..i0+23} (+8 for hi half)
  return r.v;
}

__device__ __forceinline__ v8f wmma_bf16(v16bf a, v16bf b, v8f c) {
  return __builtin_amdgcn_wmma_f32_16x16x32_bf16(false, a, false, b,
                                                 (short)0, c, false, false);
}

// ---------------------------------------------------------------------------
// 1) fp32 -> bf16 conversion (weights)
// ---------------------------------------------------------------------------
__global__ void cvt_bf16_kernel(const float* __restrict__ s,
                                unsigned short* __restrict__ d, int n) {
  int i = blockIdx.x * blockDim.x + threadIdx.x;
  if (i < n) d[i] = f2bf(s[i]);
}

// ---------------------------------------------------------------------------
// 2) offset conv: 3x3, pad 1, 18 output channels, fp32 direct
// ---------------------------------------------------------------------------
__global__ void conv_offset_kernel(const float* __restrict__ x,
                                   const float* __restrict__ w,
                                   const float* __restrict__ bias,
                                   float* __restrict__ off) {
  int idx = blockIdx.x * blockDim.x + threadIdx.x;      // B*18*HW
  if (idx >= BATCH * 18 * HW) return;
  int hw = idx % HW;
  int co = (idx / HW) % 18;
  int b  = idx / (HW * 18);
  int h = hw >> 6, wc = hw & 63;
  float acc = bias[co];
  const float* xb = x + (long)b * CIN * HW;
  const float* wr = w + (long)co * CIN * 9;
  for (int c = 0; c < CIN; ++c) {
    const float* xc = xb + c * HW;
    const float* wk = wr + c * 9;
#pragma unroll
    for (int ky = -1; ky <= 1; ++ky)
#pragma unroll
      for (int kx = -1; kx <= 1; ++kx) {
        int hh = h + ky, ww = wc + kx;
        if (hh >= 0 && hh < HDIM && ww >= 0 && ww < WDIM)
          acc += xc[hh * WDIM + ww] * wk[(ky + 1) * 3 + (kx + 1)];
      }
  }
  off[idx] = acc;
}

// ---------------------------------------------------------------------------
// 3) deformable im2col, stored TRANSPOSED: colT[b][p*RDIM + (c*9+k)] (bf16)
//    so GEMM operands read K (=r) contiguously.
// ---------------------------------------------------------------------------
__global__ void build_col_kernel(const float* __restrict__ x,
                                 const float* __restrict__ off,
                                 unsigned short* __restrict__ colT) {
  int idx = blockIdx.x * blockDim.x + threadIdx.x;      // B*9*HW
  if (idx >= BATCH * KTAPS * HW) return;
  int hw = idx % HW;
  int k  = (idx / HW) % KTAPS;
  int b  = idx / (HW * KTAPS);
  int h = hw >> 6, wc = hw & 63;

  float dy = off[((long)b * 18 + 2 * k) * HW + hw];
  float dx = off[((long)b * 18 + 2 * k + 1) * HW + hw];
  float py = (float)h + (float)(k / 3 - 1) + dy;
  float px = (float)wc + (float)(k % 3 - 1) + dx;
  float y0f = floorf(py), x0f = floorf(px);
  int y0 = (int)y0f, x0 = (int)x0f;
  float wy = py - y0f, wx = px - x0f;

  int y1 = y0 + 1, x1 = x0 + 1;
  bool vy0 = (y0 >= 0) & (y0 < HDIM), vy1 = (y1 >= 0) & (y1 < HDIM);
  bool vx0 = (x0 >= 0) & (x0 < WDIM), vx1 = (x1 >= 0) & (x1 < WDIM);
  float w00 = (vy0 & vx0) ? (1.f - wy) * (1.f - wx) : 0.f;
  float w01 = (vy0 & vx1) ? (1.f - wy) * wx         : 0.f;
  float w10 = (vy1 & vx0) ? wy * (1.f - wx)         : 0.f;
  float w11 = (vy1 & vx1) ? wy * wx                 : 0.f;
  int cy0 = min(max(y0, 0), HDIM - 1), cy1 = min(max(y1, 0), HDIM - 1);
  int cx0 = min(max(x0, 0), WDIM - 1), cx1 = min(max(x1, 0), WDIM - 1);
  int i00 = cy0 * WDIM + cx0, i01 = cy0 * WDIM + cx1;
  int i10 = cy1 * WDIM + cx0, i11 = cy1 * WDIM + cx1;

  const float* xb = x + (long)b * CIN * HW;
  unsigned short* cb = colT + (long)b * HW * RDIM + (long)hw * RDIM + k;
  for (int c = 0; c < CIN; ++c) {
    const float* xc = xb + c * HW;
    float v = w00 * xc[i00] + w01 * xc[i01] + w10 * xc[i10] + w11 * xc[i11];
    cb[c * KTAPS] = f2bf(v);
  }
}

// ---------------------------------------------------------------------------
// 4) deform-conv GEMM via WMMA, computed as outT[p, co]:
//      A = colT tile (16 p-rows x 32 r),  B = W^T tile (32 r x 16 co)
//    Both operands: 2x b128 contiguous loads. One wave computes 16 x (16*NT).
//    VTRANS=false: store bf16 natural (co,hw) layout, one b128 store/lane/tile
//    VTRANS=true : scatter into attention-transposed layout Vt[ch*HW + key]
//                  (index bijection of the (B,HW,C) reshape view).
// ---------------------------------------------------------------------------
template <int NT, bool VTRANS>
__global__ void gemm_deform_kernel(const unsigned short* __restrict__ Wbf,
                                   const unsigned short* __restrict__ colT,
                                   const float* __restrict__ bias,
                                   unsigned short* __restrict__ outbf, int Co) {
  int t = blockIdx.x;
  int pt = t % (HW / 16);          t /= (HW / 16);
  int ct = t % (Co / (16 * NT));   t /= (Co / (16 * NT));
  int b  = t;
  int lane = threadIdx.x & 31;
  int p0 = pt * 16, c0 = ct * 16 * NT;
  const unsigned short* cb = colT + (long)b * HW * RDIM;

  v8f acc[NT];
#pragma unroll
  for (int i = 0; i < NT; ++i) acc[i] = v8f{};

  for (int k0 = 0; k0 < RDIM; k0 += 32) {
    if (k0 + 32 < RDIM)   // -> global_prefetch_b8
      __builtin_prefetch(&cb[(long)(p0 + (lane & 15)) * RDIM + k0 + 32], 0, 1);
    v16bf a = load_row16(cb, RDIM, p0, k0, lane);        // A: colT[p, r]
#pragma unroll
    for (int nt = 0; nt < NT; ++nt)                      // B: W[co, r] as W^T
      acc[nt] = wmma_bf16(a, load_row16(Wbf, RDIM, c0 + nt * 16, k0, lane),
                          acc[nt]);
  }

  int n = lane & 15, half = lane >> 4;
  unsigned short* ob = outbf + (long)b * Co * HW;
#pragma unroll
  for (int nt = 0; nt < NT; ++nt) {
    int co = c0 + nt * 16 + n;           // C layout: col n per lane
    float bs = bias[co];
    if (!VTRANS) {
      BV pk;                              // rows m = j + 8*half -> p contiguous
#pragma unroll
      for (int j = 0; j < 8; ++j) pk.h[j] = f2bf(acc[nt][j] + bs);
      *(uint4*)&ob[(long)co * HW + p0 + 8 * half] = pk.q[0];
    } else {
#pragma unroll
      for (int j = 0; j < 8; ++j) {
        long f = (long)co * HW + (p0 + 8 * half + j);    // natural flat index
        ob[(long)(f & 255) * HW + (f >> 8)] = f2bf(acc[nt][j] + bs);
      }
    }
  }
}

// ---------------------------------------------------------------------------
// 5) flash-style attention + residual, 64-key chunks.
//    One wave per (batch, 16-query tile, 64-channel quarter).
//    Per chunk: 4 WMMAs for S, LDS-transpose row phase for the online softmax
//    (lane pair (r, r+16) owns row r; ONE xor-16 max + ONE xor-16 sum shuffle
//    per chunk), P packed to LDS bf16, 8 WMMAs for P*V from pre-transposed Vt.
// ---------------------------------------------------------------------------
__global__ void attention_kernel(const unsigned short* __restrict__ Qbf,
                                 const unsigned short* __restrict__ Kbf,
                                 const unsigned short* __restrict__ Vt,
                                 const float* __restrict__ x,
                                 const float* __restrict__ gamma,
                                 float* __restrict__ out) {
  int id = blockIdx.x;                 // B * 256 * 4
  int cq = id % 4;   id /= 4;
  int qt = id % (HW / 16); id /= (HW / 16);
  int b  = id;
  int lane = threadIdx.x & 31;
  int half = lane >> 4, lx = lane & 15;
  int q0 = qt * 16, chBase = cq * 64;

  const unsigned short* Q = Qbf + (long)b * C8  * HW;  // Q[i,j] = Q[i*32+j]
  const unsigned short* K = Kbf + (long)b * C8  * HW;
  const unsigned short* V = Vt  + (long)b * CIN * HW;  // Vt[ch*HW + key]

  v16bf qa = load_row16(Q, 32, q0, 0, lane);           // A: 16 queries x 32 d

  v8f acc0 = {}, acc1 = {}, acc2 = {}, acc3 = {};
  float mrun = -1e30f, lrun = 0.f;                     // state of row r = lx

  __shared__ __align__(16) float          Sf[16 * 64]; // S row-major, 4 KB
  __shared__ __align__(16) unsigned short Pb[16 * 64]; // P row-major, 2 KB
  __shared__ __align__(16) float          Al[16];      // alpha per row
  __shared__ __align__(16) float          Ls[16];      // final l per row

  for (int k0 = 0; k0 < HW; k0 += 64) {
    if (k0 + 64 < HW) {
      __builtin_prefetch(&K[(long)(k0 + 64 + lx) * 32], 0, 1);
      __builtin_prefetch(&V[(long)(chBase + lx) * HW + k0 + 64], 0, 1);
    }
    // ---- S = Q K^T for 64 keys (4 tiles) ------------------------------
    v8f s[4];
#pragma unroll
    for (int t = 0; t < 4; ++t)
      s[t] = wmma_bf16(qa, load_row16(K, 32, k0 + t * 16, 0, lane), v8f{});
    // scatter C tiles into row-major S[16][64] (lane-distinct columns)
#pragma unroll
    for (int t = 0; t < 4; ++t) {
      V8 sv; sv.v = s[t];
#pragma unroll
      for (int j = 0; j < 8; ++j)
        Sf[(8 * half + j) * 64 + t * 16 + lx] = sv.f[j];
    }
    __syncthreads();

    // ---- row phase: lane pair (r, r+16) owns row r = lx ----------------
    {
      int r = lx, cb = 32 * half;              // this lane: 32 contiguous cols
      float v[32];
      float mx = -1e30f;
#pragma unroll
      for (int g = 0; g < 8; ++g) {            // 8x ds_load_b128
        float4 t4 = *(const float4*)&Sf[r * 64 + cb + g * 4];
        v[g*4+0] = t4.x; v[g*4+1] = t4.y; v[g*4+2] = t4.z; v[g*4+3] = t4.w;
        mx = fmaxf(mx, fmaxf(fmaxf(t4.x, t4.y), fmaxf(t4.z, t4.w)));
      }
      mx = fmaxf(mx, __shfl_xor(mx, 16, 32));  // combine halves: chunk row max
      float mnew  = fmaxf(mrun, mx);
      float alpha = __expf(mrun - mnew);
      float sum = 0.f;
#pragma unroll
      for (int g = 0; g < 4; ++g) {            // exp + pack, 4x b128 stores
        BV pk;
#pragma unroll
        for (int i = 0; i < 8; ++i) {
          float p = __expf(v[g * 8 + i] - mnew);
          sum += p;
          pk.h[i] = f2bf(p);
        }
        *(uint4*)&Pb[r * 64 + cb + g * 8] = pk.q[0];
      }
      sum += __shfl_xor(sum, 16, 32);          // combine halves: chunk row sum
      lrun = lrun * alpha + sum;
      mrun = mnew;
      if (half == 0) Al[r] = alpha;            // identical on both pair lanes
    }
    __syncthreads();

    // ---- scale accumulators by alpha of their rows ---------------------
    {
      float4 a0 = *(const float4*)&Al[8 * half];
      float4 a1 = *(const float4*)&Al[8 * half + 4];
      float av[8] = {a0.x, a0.y, a0.z, a0.w, a1.x, a1.y, a1.z, a1.w};
#pragma unroll
      for (int j = 0; j < 8; ++j) {
        acc0[j] *= av[j]; acc1[j] *= av[j];
        acc2[j] *= av[j]; acc3[j] *= av[j];
      }
    }
    // ---- P*V: two A operands (keys 0-31, 32-63), 4 channel tiles -------
    v16bf pa0 = load_row16(Pb, 64, 0, 0,  lane);
    v16bf pa1 = load_row16(Pb, 64, 0, 32, lane);
#pragma unroll
    for (int t = 0; t < 4; ++t) {
      int ch = chBase + t * 16;
      v8f* ac = (t == 0) ? &acc0 : (t == 1) ? &acc1 : (t == 2) ? &acc2 : &acc3;
      *ac = wmma_bf16(pa0, load_row16(V, HW, ch, k0,      lane), *ac);
      *ac = wmma_bf16(pa1, load_row16(V, HW, ch, k0 + 32, lane), *ac);
    }
    __syncthreads();
  }

  // ---- epilogue: normalize, gamma-scale, residual ----------------------
  if (half == 0) Ls[lx] = lrun;
  __syncthreads();
  float4 l0 = *(const float4*)&Ls[8 * half];
  float4 l1 = *(const float4*)&Ls[8 * half + 4];
  float lv[8] = {l0.x, l0.y, l0.z, l0.w, l1.x, l1.y, l1.z, l1.w};

  float g = gamma[0];
  const float* xb = x   + (long)b * CIN * HW;
  float*       ob = out + (long)b * CIN * HW;
#pragma unroll
  for (int j = 0; j < 8; ++j) {
    int m = j + 8 * half;
    float inv = 1.0f / lv[j];
    long rowoff = (long)(q0 + m) * CIN;
    long f0 = rowoff + chBase + lx;
    long f1 = f0 + 16, f2 = f0 + 32, f3 = f0 + 48;
    ob[f0] = g * (acc0[j] * inv) + xb[f0];
    ob[f1] = g * (acc1[j] * inv) + xb[f1];
    ob[f2] = g * (acc2[j] * inv) + xb[f2];
    ob[f3] = g * (acc3[j] * inv) + xb[f3];
  }
}

// ---------------------------------------------------------------------------
// launcher
// ---------------------------------------------------------------------------
extern "C" void kernel_launch(void* const* d_in, const int* in_sizes, int n_in,
                              void* d_out, int out_size, void* d_ws, size_t ws_size,
                              hipStream_t stream) {
  (void)in_sizes; (void)n_in; (void)out_size; (void)ws_size;
  const float* x     = (const float*)d_in[0];
  const float* w_off = (const float*)d_in[1];
  const float* b_off = (const float*)d_in[2];
  const float* wq    = (const float*)d_in[3];
  const float* bq    = (const float*)d_in[4];
  const float* wk    = (const float*)d_in[5];
  const float* bk    = (const float*)d_in[6];
  const float* wv    = (const float*)d_in[7];
  const float* bv    = (const float*)d_in[8];
  const float* gamma = (const float*)d_in[9];
  float* out = (float*)d_out;

  // workspace carve (~45 MB total, every region 16B-aligned)
  char* ws = (char*)d_ws;
  float*          off    = (float*)(ws);                         // 589,824 B
  unsigned short* wq_bf  = (unsigned short*)(ws + 589824);       // 147,456 B
  unsigned short* wk_bf  = (unsigned short*)(ws + 737280);       // 147,456 B
  unsigned short* wv_bf  = (unsigned short*)(ws + 884736);       // 1,179,648 B
  unsigned short* col_bf = (unsigned short*)(ws + 2064384);      // 37,748,736 B
  unsigned short* q_bf   = (unsigned short*)(ws + 39813120);     // 524,288 B
  unsigned short* k_bf   = (unsigned short*)(ws + 40337408);     // 524,288 B
  unsigned short* vt_bf  = (unsigned short*)(ws + 40861696);     // 4,194,304 B

  // 1) weights -> bf16
  cvt_bf16_kernel<<<(C8 * RDIM + 255) / 256, 256, 0, stream>>>(wq, wq_bf, C8 * RDIM);
  cvt_bf16_kernel<<<(C8 * RDIM + 255) / 256, 256, 0, stream>>>(wk, wk_bf, C8 * RDIM);
  cvt_bf16_kernel<<<(CIN * RDIM + 255) / 256, 256, 0, stream>>>(wv, wv_bf, CIN * RDIM);

  // 2) offset conv
  conv_offset_kernel<<<(BATCH * 18 * HW) / 256, 256, 0, stream>>>(x, w_off, b_off, off);

  // 3) deformable im2col (bf16, transposed [p, r])
  build_col_kernel<<<(BATCH * KTAPS * HW) / 256, 256, 0, stream>>>(x, off, col_bf);

  // 4) q/k/v GEMMs on WMMA (one wave per 16x(16*NT) tile)
  gemm_deform_kernel<2, false><<<BATCH * (HW / 16) * (C8  / 32), 32, 0, stream>>>(wq_bf, col_bf, bq, q_bf, C8);
  gemm_deform_kernel<2, false><<<BATCH * (HW / 16) * (C8  / 32), 32, 0, stream>>>(wk_bf, col_bf, bk, k_bf, C8);
  gemm_deform_kernel<4, true ><<<BATCH * (HW / 16) * (CIN / 64), 32, 0, stream>>>(wv_bf, col_bf, bv, vt_bf, CIN);

  // 5) flash attention + residual (64-key chunks)
  attention_kernel<<<BATCH * (HW / 16) * 4, 32, 0, stream>>>(q_bf, k_bf, vt_bf, x, gamma, out);
}
